// BioSimulator_57526791963314
// MI455X (gfx1250) — matrix operand order repaired and checked by reading, not compiled, not run on hardware
//
#include <hip/hip_runtime.h>
#include <hip/hip_bf16.h>
#include <math.h>

typedef __attribute__((ext_vector_type(16))) _Float16 v16h;
typedef __attribute__((ext_vector_type(8)))  float    v8f;

// Problem constants (fixed by the reference)
#define NPH   1024          // phosphenes (K dimension)
#define RES   256           // output H = W
#define BATCH 2
#define KT    32            // K per WMMA
#define NKT   (NPH / KT)    // 32 K-tiles
#define NT16  (RES / 16)    // 16 row/col tiles
#define FRAG  512           // halfs per fragment (32 lanes x 16)
#define AFRAG_ELEMS (BATCH * NT16 * NKT * FRAG)   // 524288 halfs
#define BFRAG_ELEMS (BATCH * NKT * NT16 * FRAG)   // 524288 halfs

#define K_ISCALE 8e-05f
#define K_RHEO   2.39e-05f
#define K_PW     0.00017f
#define K_FREQ   300.0f
#define K_SLOPE  19152642.5f
#define K_HALF   1.057e-07f
#define K_SPREAD 0.000675f
#define K_R2S    0.5f

// ---------------------------------------------------------------------------
// Kernel 1: per-(batch, phosphene) brightness weight and 1/(2*sigma_px^2)
// ---------------------------------------------------------------------------
__global__ void bio_prep(const float* __restrict__ stim,
                         const int*   __restrict__ idx,
                         const float* __restrict__ M,
                         const float* __restrict__ px,
                         float* __restrict__ Bw,
                         float* __restrict__ inv2s2) {
    int t = blockIdx.x * blockDim.x + threadIdx.x;   // 0 .. 2047
    int b = t >> 10;
    int n = t & (NPH - 1);

    float I    = stim[b * NPH + idx[n]] * K_ISCALE;
    float Ieff = fmaxf(I - K_RHEO, 0.0f);
    float Q    = Ieff * K_PW * K_FREQ;
    float bw   = 1.0f / (1.0f + expf(-K_SLOPE * (Q - K_HALF)));

    float fov  = px[RES - 1];                 // px[0][RES-1] == xs.max() == fov
    float d2p  = (float)RES / (2.0f * fov);

    float size_base = sqrtf(I / K_SPREAD);
    float sigma     = size_base * K_R2S / M[n];
    float sigma_px  = fmaxf(sigma * d2p, 1.0f);

    Bw[t]     = bw;
    inv2s2[t] = 1.0f / (2.0f * sigma_px * sigma_px);
}

// ---------------------------------------------------------------------------
// Kernel 2: build pre-swizzled WMMA fragments.
//   Afrag[b][ht][kt][lane][16] : A = Bw * exp(-(dy*d2p)^2 * inv2s2)  (rows=h, K=n)
//   Bfrag[b][kt][wt][lane][16] : B = exp(-(dx*d2p)^2 * inv2s2)       (K=n, cols=w)
// Lane/half -> (m,k) mapping follows ISA 16-bit A 16x32 and B 32x16 layouts.
// ---------------------------------------------------------------------------
__global__ void bio_genfrag(const float* __restrict__ Bw,
                            const float* __restrict__ inv2s2,
                            const float* __restrict__ vx,
                            const float* __restrict__ vy,
                            const float* __restrict__ px,
                            const float* __restrict__ py,
                            _Float16* __restrict__ Afrag,
                            _Float16* __restrict__ Bfrag) {
    int tid = blockIdx.x * blockDim.x + threadIdx.x;   // 0 .. 2*524288-1
    float fov = px[RES - 1];
    float d2p = (float)RES / (2.0f * fov);

    bool isB = tid >= AFRAG_ELEMS;
    int  t   = isB ? tid - AFRAG_ELEMS : tid;
    int  j    = t & 15;          // half index within v16h
    int  lane = (t >> 4) & 31;
    int  tile = t >> 9;          // 0 .. 1023

    float val;
    int b;
    if (!isB) {
        // A fragment: tile = (b*16 + ht)*32 + kt
        int kt = tile & 31;
        int ht = (tile >> 5) & 15;
        b      = tile >> 9;
        int m  = lane & 15;
        int kb = (lane < 16) ? 0 : 8;
        int k  = (j < 8) ? (kb + j) : (16 + kb + (j - 8));
        int n  = kt * KT + k;
        int h  = ht * 16 + m;
        int bn = b * NPH + n;
        float dy = (py[h * RES] - vy[n]) * d2p;      // ys[h] = py[h][0]
        val = Bw[bn] * __expf(-dy * dy * inv2s2[bn]);
    } else {
        // B fragment: tile = (b*32 + kt)*16 + wt
        int wt = tile & 15;
        int kt = (tile >> 4) & 31;
        b      = tile >> 9;
        int w  = wt * 16 + (lane & 15);
        int k  = (j < 8) ? (((lane < 16) ? 0 : 8) + j) : (16 + ((lane < 16) ? 0 : 8) + (j - 8));
        int n  = kt * KT + k;
        int bn = b * NPH + n;
        float dx = (px[w] - vx[n]) * d2p;            // xs[w] = px[0][w]
        val = __expf(-dx * dx * inv2s2[bn]);
    }

    if (!isB) Afrag[t] = (_Float16)val;
    else      Bfrag[t] = (_Float16)val;
}

// ---------------------------------------------------------------------------
// Kernel 3: WMMA GEMM. One wave per 16x16 output tile; 512 waves total.
// out[b] (256x256) = clip(2 * A(256x1024) x B(1024x256), 0, 1)
// ---------------------------------------------------------------------------
__global__ void bio_gemm(const _Float16* __restrict__ Afrag,
                         const _Float16* __restrict__ Bfrag,
                         float* __restrict__ out) {
    int wave = (blockIdx.x * blockDim.x + threadIdx.x) >> 5;  // 0 .. 511
    int lane = threadIdx.x & 31;

    int wt = wave & 15;
    int ht = (wave >> 4) & 15;
    int b  = wave >> 8;

    const _Float16* Ab = Afrag + ((size_t)(b * NT16 + ht) * NKT) * FRAG + lane * 16;
    const _Float16* Bb = Bfrag + (((size_t)(b * NKT) * NT16) + wt) * FRAG + lane * 16;

    v8f acc = {};
#pragma unroll 4
    for (int kt = 0; kt < NKT; ++kt) {
        v16h a  = *(const v16h*)(Ab + (size_t)kt * FRAG);
        v16h bf = *(const v16h*)(Bb + (size_t)kt * NT16 * FRAG);
        acc = __builtin_amdgcn_wmma_f32_16x16x32_f16(
            /*neg_a=*/false, a, /*neg_b=*/false, bf,
            /*c_mod=*/(short)0, acc, /*reuse_a=*/false, /*reuse_b=*/false);
    }

    // C/D layout: VGPR r -> row r (lanes 0-15) or row r+8 (lanes 16-31); lane&15 -> col
    int col  = wt * 16 + (lane & 15);
    int row0 = ht * 16 + ((lane < 16) ? 0 : 8);
    float* o = out + (size_t)b * RES * RES;
#pragma unroll
    for (int r = 0; r < 8; ++r) {
        float v = acc[r] * 2.0f;
        v = fminf(fmaxf(v, 0.0f), 1.0f);
        o[(size_t)(row0 + r) * RES + col] = v;
    }
}

// ---------------------------------------------------------------------------
// Launch
// ---------------------------------------------------------------------------
extern "C" void kernel_launch(void* const* d_in, const int* in_sizes, int n_in,
                              void* d_out, int out_size, void* d_ws, size_t ws_size,
                              hipStream_t stream) {
    const float* stim = (const float*)d_in[0];   // [2,32,32]
    const float* vx   = (const float*)d_in[1];   // [1024]
    const float* vy   = (const float*)d_in[2];   // [1024]
    const float* M    = (const float*)d_in[3];   // [1024]
    const float* px   = (const float*)d_in[4];   // [256,256]
    const float* py   = (const float*)d_in[5];   // [256,256]
    const int*   idx  = (const int*)d_in[6];     // [1024]
    float* out = (float*)d_out;                  // [2,1,256,256]

    // Workspace layout: Bw(2048 f32) | inv2s2(2048 f32) | Afrag(512K f16) | Bfrag(512K f16)
    float*    Bw     = (float*)d_ws;
    float*    inv2s2 = Bw + BATCH * NPH;
    _Float16* Afrag  = (_Float16*)(inv2s2 + BATCH * NPH);
    _Float16* Bfrag  = Afrag + AFRAG_ELEMS;

    bio_prep<<<(BATCH * NPH) / 256, 256, 0, stream>>>(stim, idx, M, px, Bw, inv2s2);

    int genThreads = AFRAG_ELEMS + BFRAG_ELEMS;          // 1,048,576
    bio_genfrag<<<genThreads / 256, 256, 0, stream>>>(Bw, inv2s2, vx, vy, px, py,
                                                      Afrag, Bfrag);

    // 512 waves = 64 blocks x 8 waves; every wave has full EXEC (WMMA requirement)
    bio_gemm<<<64, 256, 0, stream>>>(Afrag, Bfrag, out);
}
